// Model_76527727280465
// MI455X (gfx1250) — matrix operand order repaired and checked by reading, not compiled, not run on hardware
//
#include <hip/hip_runtime.h>
#include <hip/hip_bf16.h>
#include <math.h>

typedef __attribute__((ext_vector_type(16))) __bf16          v16bf;
typedef __attribute__((ext_vector_type(16))) unsigned short  v16us;
typedef __attribute__((ext_vector_type(8)))  float           v8f;

#define DMODEL 512
#define NHEADS 8
#define DH     64
#define NHASH  4
#define BUCKET 24

#define CDIV(a,b) (((a)+(b)-1)/(b))

// ---------------------------------------------------------------------------
// helpers
// ---------------------------------------------------------------------------
static __device__ __forceinline__ unsigned short f2bf(float f) {
  unsigned u = __float_as_uint(f);
  u = (u + 0x7FFFu + ((u >> 16) & 1u)) >> 16;   // RNE fp32 -> bf16
  return (unsigned short)u;
}

static __device__ __forceinline__ unsigned pack2bf(float lo, float hi) {
  return (unsigned)f2bf(lo) | ((unsigned)f2bf(hi) << 16);
}

static __device__ __forceinline__ float gelu_f(float x) {
  return 0.5f * x * (1.f + tanhf(0.7978845608f * (x + 0.044715f * x * x * x)));
}

// CDNA5 async global->LDS DMA (ASYNCcnt path); 16B per lane per issue.
static __device__ __forceinline__ void async_load_b128(void* lds_dst, const void* gsrc) {
  unsigned ldsa = (unsigned)(uintptr_t)lds_dst;   // generic LDS addr[31:0] == LDS offset
  asm volatile("global_load_async_to_lds_b128 %0, %1, off"
               :: "v"(ldsa), "v"(gsrc) : "memory");
}
static __device__ __forceinline__ void wait_async0() {
  asm volatile("s_wait_asynccnt 0x0" ::: "memory");
}

// ---------------------------------------------------------------------------
// weight pre-pass: f32 (K,N) row-major  ->  bf16 [N][K] (transposed)
// ---------------------------------------------------------------------------
__global__ void wconv_kernel(const float* __restrict__ w, unsigned short* __restrict__ wt,
                             int K, int N) {
  int idx = blockIdx.x * blockDim.x + threadIdx.x;
  if (idx >= N * K) return;
  int n = idx / K, k = idx % K;
  wt[idx] = f2bf(w[(size_t)k * N + n]);
}

// ---------------------------------------------------------------------------
// WMMA GEMM:  C[M,N] = act(A[M,K] @ W[K,N] + bias) + residual
// Double-buffered LDS pipeline:
//   B tile: bf16 [N][K] weights staged via global_load_async_to_lds_b128 (DMA)
//   A tile: f32 vector loads -> packed bf16 -> ds_store_b128
//   tile k+1 DMA/loads issued before tile k WMMAs -> latency hidden by math.
// 64x64 block tile, K-step 32, 8 waves; each wave: 16x32 out (2 WMMA accums).
// M, N, K multiples of 64/64/32 (true for every call in this model).
// ---------------------------------------------------------------------------
#define TM 64
#define TN 64
#define TK 32

template<bool HAS_BIAS, bool HAS_RES, bool ACT_GELU>
__global__ __launch_bounds__(256) void gemm_wmma_kernel(
    const float* __restrict__ A, const unsigned short* __restrict__ Bt,
    float* __restrict__ C, int M, int N, int K,
    const float* __restrict__ bias, const float* __restrict__ residual)
{
  __shared__ __align__(16) unsigned short As[2][TM][TK];   // [buf][m][k] bf16
  __shared__ __align__(16) unsigned short Bs[2][TN][TK];   // [buf][n][k] bf16

  const int tid  = threadIdx.x;
  const int lane = tid & 31;
  const int wave = tid >> 5;
  const int m0   = blockIdx.y * TM;
  const int n0   = blockIdx.x * TN;

  const int mtile  = wave & 3;        // 4 m-tiles of 16
  const int ntile0 = (wave >> 2) * 2; // 2 n-tiles of 16 per wave

  v8f accs[2];
  accs[0] = (v8f){0,0,0,0,0,0,0,0};
  accs[1] = (v8f){0,0,0,0,0,0,0,0};

  const int half = lane >> 4;
  const int l16  = lane & 15;

  // staging coordinates: thread i handles 8 contiguous k-elements of one row
  const int sm = tid >> 2;          // 0..63
  const int sk = (tid & 3) * 8;     // 0,8,16,24

  const unsigned short* gB = Bt + (size_t)(n0 + sm) * K + sk;
  const float*          gA = A  + (size_t)(m0 + sm) * K + sk;

  // ---- prologue: stage tile 0 into buffer 0 ----
  async_load_b128(&Bs[0][sm][sk], gB);
  {
    float4 f0 = *(const float4*)gA;
    float4 f1 = *(const float4*)(gA + 4);
    uint4 pk;
    pk.x = pack2bf(f0.x, f0.y);
    pk.y = pack2bf(f0.z, f0.w);
    pk.z = pack2bf(f1.x, f1.y);
    pk.w = pack2bf(f1.z, f1.w);
    *(uint4*)&As[0][sm][sk] = pk;
  }
  wait_async0();
  __syncthreads();

  const int nsteps = K / TK;
  for (int step = 0; step < nsteps; ++step) {
    const int buf = step & 1;
    const int nxt = buf ^ 1;
    const bool more = (step + 1) < nsteps;

    // ---- issue next tile's DMA + global loads before this tile's math ----
    float4 f0 = {0,0,0,0}, f1 = {0,0,0,0};
    if (more) {
      const int k1 = (step + 1) * TK;
      async_load_b128(&Bs[nxt][sm][sk], gB + k1);
      f0 = *(const float4*)(gA + k1);
      f1 = *(const float4*)(gA + k1 + 4);
    }

    // ---- fragments + WMMA from current buffer ----
    const int mrow = mtile * 16 + l16;
    v16us au;
    #pragma unroll
    for (int v = 0; v < 8; ++v) {
      int kb = ((v < 4) ? 0 : 16) + 8 * half + 2 * (v & 3);
      au[2*v]   = As[buf][mrow][kb];
      au[2*v+1] = As[buf][mrow][kb+1];
    }
    v16bf af = __builtin_bit_cast(v16bf, au);

    #pragma unroll
    for (int t = 0; t < 2; ++t) {
      const int ncol = (ntile0 + t) * 16 + l16;
      v16us bu;
      #pragma unroll
      for (int v = 0; v < 8; ++v) {
        int kb = ((v < 4) ? 0 : 16) + 8 * half + 2 * (v & 3);
        bu[2*v]   = Bs[buf][ncol][kb];
        bu[2*v+1] = Bs[buf][ncol][kb+1];
      }
      v16bf bf = __builtin_bit_cast(v16bf, bu);
      accs[t] = __builtin_amdgcn_wmma_f32_16x16x32_bf16(
          false, af, false, bf, (short)0, accs[t], false, false);
    }

    // ---- finish staging next buffer, then rendezvous ----
    if (more) {
      uint4 pk;
      pk.x = pack2bf(f0.x, f0.y);
      pk.y = pack2bf(f0.z, f0.w);
      pk.z = pack2bf(f1.x, f1.y);
      pk.w = pack2bf(f1.z, f1.w);
      *(uint4*)&As[nxt][sm][sk] = pk;
      wait_async0();
    }
    __syncthreads();
  }

  // epilogue: C/D layout -> lane(n = l%16), vgpr r -> m = 8*half + r
  #pragma unroll
  for (int t = 0; t < 2; ++t) {
    int gn = n0 + (ntile0 + t) * 16 + l16;
    float bb = HAS_BIAS ? bias[gn] : 0.f;
    #pragma unroll
    for (int r = 0; r < 8; ++r) {
      int gm = m0 + mtile * 16 + half * 8 + r;
      float v = accs[t][r] + bb;
      if (ACT_GELU) v = gelu_f(v);
      if (HAS_RES)  v += residual[(size_t)gm * N + gn];
      C[(size_t)gm * N + gn] = v;
    }
  }
}

// host-side dispatch over epilogue variants
static inline void launch_gemm(dim3 grid, hipStream_t st,
                               const float* A, const unsigned short* Bt, float* C,
                               int M, int N, int K,
                               const float* bias, const float* res, int act) {
  if (act)
    gemm_wmma_kernel<true,  false, true ><<<grid, 256, 0, st>>>(A, Bt, C, M, N, K, bias, res);
  else if (bias && res)
    gemm_wmma_kernel<true,  true,  false><<<grid, 256, 0, st>>>(A, Bt, C, M, N, K, bias, res);
  else if (res)
    gemm_wmma_kernel<false, true,  false><<<grid, 256, 0, st>>>(A, Bt, C, M, N, K, bias, res);
  else
    gemm_wmma_kernel<false, false, false><<<grid, 256, 0, st>>>(A, Bt, C, M, N, K, bias, res);
}

// ---------------------------------------------------------------------------
// LayerNorm over D=512, one block per row
// ---------------------------------------------------------------------------
__global__ __launch_bounds__(256) void layernorm_kernel(
    const float* __restrict__ x, const float* __restrict__ g,
    const float* __restrict__ b, float* __restrict__ out, int D)
{
  __shared__ float red[256];
  const int row = blockIdx.x;
  const float* xr = x + (size_t)row * D;

  float s = 0.f;
  for (int i = threadIdx.x; i < D; i += 256) s += xr[i];
  red[threadIdx.x] = s; __syncthreads();
  for (int off = 128; off > 0; off >>= 1) {
    if (threadIdx.x < off) red[threadIdx.x] += red[threadIdx.x + off];
    __syncthreads();
  }
  float mu = red[0] / D;
  __syncthreads();

  float v = 0.f;
  for (int i = threadIdx.x; i < D; i += 256) { float d = xr[i] - mu; v += d * d; }
  red[threadIdx.x] = v; __syncthreads();
  for (int off = 128; off > 0; off >>= 1) {
    if (threadIdx.x < off) red[threadIdx.x] += red[threadIdx.x + off];
    __syncthreads();
  }
  float inv = rsqrtf(red[0] / D + 1e-5f);

  for (int i = threadIdx.x; i < D; i += 256)
    out[(size_t)row * D + i] = (xr[i] - mu) * inv * g[i] + b[i];
}

// ---------------------------------------------------------------------------
// small elementwise kernels
// ---------------------------------------------------------------------------
__global__ void zero_kernel(float* p, long n) {
  long i = (long)blockIdx.x * blockDim.x + threadIdx.x;
  if (i < n) p[i] = 0.f;
}

__global__ void build_mark_kernel(const float* __restrict__ me,
                                  const float* __restrict__ md,
                                  float* __restrict__ mark) {
  int idx = blockIdx.x * blockDim.x + threadIdx.x;
  const int total = 32 * 576 * 4;
  if (idx >= total) return;
  int c = idx & 3, l = (idx >> 2) % 576, b = idx / (576 * 4);
  float v = (l < 384) ? me[((size_t)b * 384 + l) * 4 + c]
                      : md[((size_t)b * 384 + (l - 192)) * 4 + c];
  mark[idx] = v;
}

// moving average with separate output batch stride (for partial overwrites)
__global__ void movavg_kernel(const float* __restrict__ x, float* __restrict__ out,
                              int B, int Lin, int C, int s, int Lout, int ostride) {
  int idx = blockIdx.x * blockDim.x + threadIdx.x;
  int total = B * Lout * C;
  if (idx >= total) return;
  int c = idx % C, lo = (idx / C) % Lout, b = idx / (C * Lout);
  float acc = 0.f;
  for (int i = 0; i < s; ++i) acc += x[((size_t)b * Lin + lo * s + i) * C + c];
  out[((size_t)b * ostride + lo) * C + c] = acc / s;
}

__global__ void upsample_kernel(const float* __restrict__ x, float* __restrict__ out,
                                int B, int Lin, int C) {
  int Lo = Lin * 4;
  int idx = blockIdx.x * blockDim.x + threadIdx.x;
  int total = B * Lo * C;
  if (idx >= total) return;
  int c = idx % C, lo = (idx / C) % Lo, b = idx / (C * Lo);
  float pos  = (lo + 0.5f) / 4.f - 0.5f;
  float fl   = floorf(pos);
  float frac = pos - fl;
  int loi = (int)fl; if (loi < 0) loi = 0; if (loi > Lin - 1) loi = Lin - 1;
  int hii = loi + 1; if (hii > Lin - 1) hii = Lin - 1;
  if (pos < 0.f || pos > (float)(Lin - 1)) frac = 0.f;
  float xl = x[((size_t)b * Lin + loi) * C + c];
  float xh = x[((size_t)b * Lin + hii) * C + c];
  out[idx] = xl * (1.f - frac) + xh * frac;
}

__global__ void mean_kernel(const float* __restrict__ enc, const float* __restrict__ dec,
                            float* __restrict__ mean, int B, int Le, int Ld, int tail) {
  int idx = blockIdx.x * blockDim.x + threadIdx.x;
  if (idx >= B * 7) return;
  int c = idx % 7, b = idx / 7;
  float acc = 0.f;
  for (int l = 0; l < Le; ++l)  acc += enc[((size_t)b * Le + l) * 7 + c];
  for (int j = 0; j < tail; ++j) acc += dec[((size_t)b * Ld + (Ld - tail) + j) * 7 + c];
  mean[idx] = acc / (float)(Le + tail);
}

__global__ void build_cat_kernel(const float* __restrict__ enc, const float* __restrict__ dec,
                                 const float* __restrict__ mean, float* __restrict__ out,
                                 int B, int Le, int Ld, int tail) {
  int Ls = Le + tail;
  int idx = blockIdx.x * blockDim.x + threadIdx.x;
  int total = B * Ls * 7;
  if (idx >= total) return;
  int c = idx % 7, l = (idx / 7) % Ls, b = idx / (7 * Ls);
  float m = mean[b * 7 + c];
  float v = (l < Le) ? enc[((size_t)b * Le + l) * 7 + c]
                     : dec[((size_t)b * Ld + (Ld - tail) + (l - Le)) * 7 + c];
  out[idx] = v - m;
}

__global__ void embed_kernel(const float* __restrict__ xc, const float* __restrict__ mark,
                             const float* __restrict__ convw, const float* __restrict__ tempw,
                             const float* __restrict__ scalew, float* __restrict__ out,
                             int B, int Ls, int scale, int si) {
  int idx = blockIdx.x * blockDim.x + threadIdx.x;
  int total = B * Ls * DMODEL;
  if (idx >= total) return;
  int d = idx & (DMODEL - 1), l = (idx >> 9) % Ls, b = idx / (DMODEL * Ls);
  float acc = scalew[si * DMODEL + d];
  #pragma unroll
  for (int kw = 0; kw < 3; ++kw) {
    int xl = l + kw - 1;
    if (xl < 0)   xl = Ls - 1;
    if (xl >= Ls) xl = 0;
    const float* xr = xc + ((size_t)b * Ls + xl) * 7;
    const float* wr = convw + (size_t)(kw * 7) * DMODEL + d;
    #pragma unroll
    for (int ci = 0; ci < 7; ++ci) acc += xr[ci] * wr[(size_t)ci * DMODEL];
  }
  int ml = scale / 2 + l * scale;
  const float* mr = mark + ((size_t)b * 576 + ml) * 4;
  #pragma unroll
  for (int t = 0; t < 4; ++t) acc += mr[t] * tempw[t * DMODEL + d];
  out[idx] = acc;
}

// ---------------------------------------------------------------------------
// LSH attention pieces
// ---------------------------------------------------------------------------
__global__ void rot_init_kernel(float* rot, int n, unsigned seed) {
  int i = blockIdx.x * blockDim.x + threadIdx.x;
  if (i >= n) return;
  unsigned s = seed + (unsigned)i * 0x9E3779B9u;
  s ^= s >> 16; s *= 0x7FEB352Du; s ^= s >> 15; s *= 0x846CA68Bu; s ^= s >> 16;
  unsigned t = s + 0x632BE5ABu;
  t ^= t >> 16; t *= 0x7FEB352Du; t ^= t >> 15; t *= 0x846CA68Bu; t ^= t >> 16;
  float u1 = (float)((s >> 8) + 1u) * (1.0f / 16777217.f);
  float u2 = (float)(t >> 8) * (1.0f / 16777216.f);
  rot[i] = sqrtf(-2.f * logf(u1)) * cosf(6.28318530718f * u2);
}

__global__ void lsh_bucket_kernel(const float* __restrict__ qk, const float* __restrict__ rot,
                                  int* __restrict__ buckets, int* __restrict__ skey,
                                  int B, int H, int R, int L, int nb2) {
  int idx = blockIdx.x * blockDim.x + threadIdx.x;
  int total = B * H * R * L;
  if (idx >= total) return;
  int l = idx % L, r = (idx / L) % R, h = (idx / (L * R)) % H, b = idx / (L * R * H);
  const float* q = qk + (((size_t)b * L + l) * H + h) * DH;
  float accs[16];
  for (int k = 0; k < nb2; ++k) {
    float a = 0.f;
    #pragma unroll
    for (int d = 0; d < DH; ++d) a += q[d] * rot[((size_t)r * DH + d) * nb2 + k];
    accs[k] = a;
  }
  float best = -1e30f; int bi = 0;
  for (int k = 0; k < nb2; ++k) if (accs[k]  > best) { best = accs[k];  bi = k; }
  for (int k = 0; k < nb2; ++k) if (-accs[k] > best) { best = -accs[k]; bi = nb2 + k; }
  int nb = 2 * nb2;
  int bucket = bi + r * nb;
  buckets[idx] = bucket;
  skey[idx]    = bucket * L + l;
}

__global__ __launch_bounds__(256) void bitonic_sort_kernel(
    const int* __restrict__ skey, int* __restrict__ ticker, int* __restrict__ undo,
    int L, int npow2) {
  __shared__ int key[1024];
  __shared__ int idx[1024];
  int grp = blockIdx.x;
  const int* sk = skey + (size_t)grp * L;
  for (int i = threadIdx.x; i < npow2; i += blockDim.x) {
    key[i] = (i < L) ? sk[i] : 0x7FFFFFFF;
    idx[i] = i;
  }
  __syncthreads();
  for (int k = 2; k <= npow2; k <<= 1) {
    for (int j = k >> 1; j > 0; j >>= 1) {
      for (int i = threadIdx.x; i < npow2; i += blockDim.x) {
        int p = i ^ j;
        if (p > i) {
          bool up = ((i & k) == 0);
          int ki = key[i], kp = key[p];
          if ((ki > kp) == up) {
            key[i] = kp; key[p] = ki;
            int t = idx[i]; idx[i] = idx[p]; idx[p] = t;
          }
        }
      }
      __syncthreads();
    }
  }
  for (int i = threadIdx.x; i < L; i += blockDim.x) {
    int org = idx[i];
    ticker[(size_t)grp * L + i] = org;
    undo[(size_t)grp * L + org] = i;
  }
}

__global__ __launch_bounds__(64) void lsh_logits_kernel(
    const float* __restrict__ qk, const int* __restrict__ ticker,
    const int* __restrict__ buckets, float* __restrict__ logits,
    int B, int H, int R, int L, int chunk, int nchunks) {
  __shared__ float kk[72][DH];
  __shared__ float rnorm[72];
  __shared__ int   kb[72];
  __shared__ int   kpos[72];

  int blk = blockIdx.x;
  int c = blk % nchunks;
  int r = (blk / nchunks) % R;
  int h = (blk / (nchunks * R)) % H;
  int b =  blk / (nchunks * R * H);
  size_t base = (((size_t)b * H + h) * R + r) * L;
  int nk = 2 * chunk;

  for (int j = threadIdx.x; j < nk; j += blockDim.x) {
    int src_c = (j < chunk) ? c : (c - 1 + nchunks) % nchunks;
    int p = src_c * chunk + (j % chunk);
    int lorig = ticker[base + p];
    kpos[j] = lorig;
    kb[j]   = buckets[base + lorig];
    const float* krow = qk + (((size_t)b * L + lorig) * H + h) * DH;
    float nrm = 0.f;
    #pragma unroll
    for (int d = 0; d < DH; ++d) { float x = krow[d]; kk[j][d] = x; nrm += x * x; }
    rnorm[j] = 1.f / (sqrtf(nrm) + 1e-6f);
  }
  __syncthreads();

  int i = threadIdx.x;
  if (i < chunk) {
    int p  = c * chunk + i;
    int li = ticker[base + p];
    int qb = buckets[base + li];
    const float* qrow = qk + (((size_t)b * L + li) * H + h) * DH;
    float q[DH];
    #pragma unroll
    for (int d = 0; d < DH; ++d) q[d] = qrow[d];
    float m = -1e30f, ssum = 0.f;
    for (int j = 0; j < nk; ++j) {
      float a = 0.f;
      #pragma unroll
      for (int d = 0; d < DH; ++d) a += q[d] * kk[j][d];
      a *= rnorm[j] * 0.125f;
      if (qb != kb[j])   a = -1e9f;
      if (li == kpos[j]) a = -1e5f;
      if (a > m) { ssum = ssum * expf(m - a) + 1.f; m = a; }
      else       { ssum += expf(a - m); }
    }
    logits[base + p] = m + logf(ssum);
  }
}

__global__ void merge_weights_kernel(const float* __restrict__ logits,
                                     const int* __restrict__ undo,
                                     float* __restrict__ wsm,
                                     int total, int R, int L) {
  int idx = blockIdx.x * blockDim.x + threadIdx.x;
  if (idx >= total) return;
  int l = idx % L, bh = idx / L;
  float lg[8];
  float mx = -1e30f;
  for (int r = 0; r < R; ++r) {
    size_t base = ((size_t)bh * R + r) * L;
    lg[r] = logits[base + undo[base + l]];
    if (lg[r] > mx) mx = lg[r];
  }
  float s = 0.f;
  for (int r = 0; r < R; ++r) { lg[r] = expf(lg[r] - mx); s += lg[r]; }
  for (int r = 0; r < R; ++r)
    wsm[((size_t)bh * R + r) * L + l] = lg[r] / s;
}

__global__ __launch_bounds__(64) void lsh_output_kernel(
    const float* __restrict__ qk, const float* __restrict__ vbuf,
    const int* __restrict__ ticker, const int* __restrict__ buckets,
    const float* __restrict__ logits, const float* __restrict__ wsm,
    float* __restrict__ attn,
    int B, int H, int R, int L, int chunk, int nchunks) {
  __shared__ float kk[72][DH];
  __shared__ float vv[72][DH];
  __shared__ float rnorm[72];
  __shared__ int   kb[72];
  __shared__ int   kpos[72];

  int blk = blockIdx.x;
  int c = blk % nchunks;
  int r = (blk / nchunks) % R;
  int h = (blk / (nchunks * R)) % H;
  int b =  blk / (nchunks * R * H);
  size_t base = (((size_t)b * H + h) * R + r) * L;
  int nk = 2 * chunk;

  for (int j = threadIdx.x; j < nk; j += blockDim.x) {
    int src_c = (j < chunk) ? c : (c - 1 + nchunks) % nchunks;
    int p = src_c * chunk + (j % chunk);
    int lorig = ticker[base + p];
    kpos[j] = lorig;
    kb[j]   = buckets[base + lorig];
    const float* krow = qk   + (((size_t)b * L + lorig) * H + h) * DH;
    const float* vrow = vbuf + (((size_t)b * L + lorig) * H + h) * DH;
    float nrm = 0.f;
    #pragma unroll
    for (int d = 0; d < DH; ++d) {
      float x = krow[d]; kk[j][d] = x; nrm += x * x;
      vv[j][d] = vrow[d];
    }
    rnorm[j] = 1.f / (sqrtf(nrm) + 1e-6f);
  }
  __syncthreads();

  int i = threadIdx.x;
  if (i < chunk) {
    int p  = c * chunk + i;
    int li = ticker[base + p];
    int qb = buckets[base + li];
    const float* qrow = qk + (((size_t)b * L + li) * H + h) * DH;
    float q[DH];
    #pragma unroll
    for (int d = 0; d < DH; ++d) q[d] = qrow[d];
    float lg = logits[base + p];
    float w  = wsm[base + li];
    float acc[DH];
    #pragma unroll
    for (int d = 0; d < DH; ++d) acc[d] = 0.f;
    for (int j = 0; j < nk; ++j) {
      float a = 0.f;
      #pragma unroll
      for (int d = 0; d < DH; ++d) a += q[d] * kk[j][d];
      a *= rnorm[j] * 0.125f;
      if (qb != kb[j])   a = -1e9f;
      if (li == kpos[j]) a = -1e5f;
      float pr = expf(a - lg);
      #pragma unroll
      for (int d = 0; d < DH; ++d) acc[d] += pr * vv[j][d];
    }
    float* dst = attn + (((size_t)b * L + li) * H + h) * DH;
    #pragma unroll
    for (int d = 0; d < DH; ++d) atomicAdd(&dst[d], w * acc[d]);
  }
}

// ---------------------------------------------------------------------------
// final projection (+bias +mean) and row-slice copy
// ---------------------------------------------------------------------------
__global__ void proj_kernel(const float* __restrict__ h, const float* __restrict__ w,
                            const float* __restrict__ bias, const float* __restrict__ mean,
                            float* __restrict__ hproj, int B, int Ls) {
  int idx = blockIdx.x * blockDim.x + threadIdx.x;
  int total = B * Ls * 7;
  if (idx >= total) return;
  int c = idx % 7, l = (idx / 7) % Ls, b = idx / (7 * Ls);
  const float* hr = h + ((size_t)b * Ls + l) * DMODEL;
  float acc = bias[c];
  for (int k = 0; k < DMODEL; ++k) acc += hr[k] * w[(size_t)k * 7 + c];
  hproj[idx] = acc + mean[b * 7 + c];
}

__global__ void copy_rows_kernel(const float* __restrict__ src, float* __restrict__ dst,
                                 int B, int Lsrc, int n, int start) {
  int idx = blockIdx.x * blockDim.x + threadIdx.x;
  int total = B * n * 7;
  if (idx >= total) return;
  int c = idx % 7, t = (idx / 7) % n, b = idx / (7 * n);
  dst[idx] = src[((size_t)b * Lsrc + start + t) * 7 + c];
}

// ---------------------------------------------------------------------------
// orchestration
// ---------------------------------------------------------------------------
extern "C" void kernel_launch(void* const* d_in, const int* in_sizes, int n_in,
                              void* d_out, int out_size, void* d_ws, size_t ws_size,
                              hipStream_t stream) {
  (void)in_sizes; (void)n_in; (void)out_size; (void)ws_size;

  const float* x_enc      = (const float*)d_in[0];
  const float* x_mark_enc = (const float*)d_in[1];
  const float* x_dec      = (const float*)d_in[2];
  const float* x_mark_dec = (const float*)d_in[3];
  int pi = 4;
  const float* emb_conv_w  = (const float*)d_in[pi++];
  const float* emb_temp_w  = (const float*)d_in[pi++];
  const float* emb_scale_w = (const float*)d_in[pi++];
  struct Layer { const float *wqk,*wv,*wo,*ff1w,*ff1b,*ff2w,*ff2b,*ln1g,*ln1b,*ln2g,*ln2b; } Lp[2];
  for (int l = 0; l < 2; ++l) {
    Lp[l].wqk  = (const float*)d_in[pi++];
    Lp[l].wv   = (const float*)d_in[pi++];
    Lp[l].wo   = (const float*)d_in[pi++];
    Lp[l].ff1w = (const float*)d_in[pi++];
    Lp[l].ff1b = (const float*)d_in[pi++];
    Lp[l].ff2w = (const float*)d_in[pi++];
    Lp[l].ff2b = (const float*)d_in[pi++];
    Lp[l].ln1g = (const float*)d_in[pi++];
    Lp[l].ln1b = (const float*)d_in[pi++];
    Lp[l].ln2g = (const float*)d_in[pi++];
    Lp[l].ln2b = (const float*)d_in[pi++];
  }
  const float* final_g = (const float*)d_in[pi++];
  const float* final_b = (const float*)d_in[pi++];
  const float* proj_w  = (const float*)d_in[pi++];
  const float* proj_b  = (const float*)d_in[pi++];

  float* out = (float*)d_out;
  float* ws  = (float*)d_ws;

  const size_t MAXM   = 32 * 576;
  const size_t SZ_MAT = MAXM * DMODEL;       // 9,437,184 floats
  float* hbuf  = ws;
  float* tmp   = ws + SZ_MAT;
  float* big   = ws + 2 * SZ_MAT;            // 4*SZ_MAT region
  float* qkb   = big;
  float* vb    = big + SZ_MAT;
  float* attnb = big + 2 * SZ_MAT;
  float* ffmid = big;                         // aliases qk/v/attn (+1 extra slab)
  float* misc  = ws + 6 * SZ_MAT;

  float* mark   = misc;                 misc += 32 * 576 * 4;
  float* enc_o  = misc;                 misc += 32 * 384 * 7;
  float* dec_o  = misc;                 misc += 32 * 384 * 7;
  float* catb   = misc;                 misc += 32 * 576 * 7;
  float* meanb  = misc;                 misc += 256;
  float* coarse = misc;                 misc += 32 * 384 * 7;
  float* hproj  = misc;                 misc += 32 * 576 * 7;
  float* rot    = misc;                 misc += NHASH * DH * 16;
  const size_t LSHN = (size_t)32 * NHEADS * NHASH * 576;
  int*   buckets = (int*)misc;          misc += LSHN;
  int*   skey    = (int*)misc;          misc += LSHN;
  int*   ticker  = (int*)misc;          misc += LSHN;
  int*   undo    = (int*)misc;          misc += LSHN;
  float* logits  = misc;                misc += LSHN;
  float* wsm     = misc;                misc += LSHN;

  // bf16 transposed weights [N][K]
  unsigned short* wb = (unsigned short*)misc;
  struct LayerB { unsigned short *wqk,*wv,*wo,*ff1,*ff2; } Lb[2];
  {
    unsigned short* p = wb;
    for (int l = 0; l < 2; ++l) {
      Lb[l].wqk = p; p += 512 * 512;
      Lb[l].wv  = p; p += 512 * 512;
      Lb[l].wo  = p; p += 512 * 512;
      Lb[l].ff1 = p; p += 2048 * 512;   // [N=2048][K=512]
      Lb[l].ff2 = p; p += 512 * 2048;   // [N=512][K=2048]
    }
  }

  // one-time weight convert+transpose (stream-ordered before first GEMM)
  for (int l = 0; l < 2; ++l) {
    wconv_kernel<<<CDIV(512 * 512, 256), 256, 0, stream>>>(Lp[l].wqk,  Lb[l].wqk, 512, 512);
    wconv_kernel<<<CDIV(512 * 512, 256), 256, 0, stream>>>(Lp[l].wv,   Lb[l].wv,  512, 512);
    wconv_kernel<<<CDIV(512 * 512, 256), 256, 0, stream>>>(Lp[l].wo,   Lb[l].wo,  512, 512);
    wconv_kernel<<<CDIV(512 * 2048, 256), 256, 0, stream>>>(Lp[l].ff1w, Lb[l].ff1, 512, 2048);
    wconv_kernel<<<CDIV(2048 * 512, 256), 256, 0, stream>>>(Lp[l].ff2w, Lb[l].ff2, 2048, 512);
  }

  const int scales[3] = {16, 4, 1};
  const int outoff[3] = {0, 32 * 12 * 7, 32 * 12 * 7 + 32 * 48 * 7};

  build_mark_kernel<<<CDIV(32 * 576 * 4, 256), 256, 0, stream>>>(x_mark_enc, x_mark_dec, mark);

  for (int si = 0; si < 3; ++si) {
    const int s    = scales[si];
    const int Le   = 384 / s;
    const int Ld   = 384 / s;
    const int tail = 192 / s;
    const int Ls   = 576 / s;
    const int M    = 32 * Ls;

    movavg_kernel<<<CDIV(32 * Le * 7, 256), 256, 0, stream>>>(x_enc, enc_o, 32, 384, 7, s, Le, Le);

    if (si == 0) {
      movavg_kernel<<<CDIV(32 * Ld * 7, 256), 256, 0, stream>>>(x_dec, dec_o, 32, 384, 7, s, Ld, Ld);
    } else {
      const int Lprev = 96 / s;          // 384 / (4*s)
      upsample_kernel<<<CDIV(32 * Ld * 7, 256), 256, 0, stream>>>(coarse, dec_o, 32, Lprev, 7);
      movavg_kernel<<<CDIV(32 * tail * 7, 256), 256, 0, stream>>>(x_dec, dec_o, 32, 384, 7, s, tail, Ld);
    }

    mean_kernel<<<CDIV(32 * 7, 128), 128, 0, stream>>>(enc_o, dec_o, meanb, 32, Le, Ld, tail);
    build_cat_kernel<<<CDIV(32 * Ls * 7, 256), 256, 0, stream>>>(enc_o, dec_o, meanb, catb, 32, Le, Ld, tail);

    embed_kernel<<<CDIV(M * DMODEL, 256), 256, 0, stream>>>(catb, mark, emb_conv_w, emb_temp_w,
                                                            emb_scale_w, hbuf, 32, Ls, s, si);

    int nb = Ls / BUCKET; if (nb < 2) nb = 2; if (nb & 1) nb += 1;
    const int nb2 = nb / 2;
    const int chunk   = (Ls % BUCKET == 0) ? BUCKET : Ls;
    const int nchunks = (Ls % BUCKET == 0) ? Ls / BUCKET : 1;
    int npow2 = 1; while (npow2 < Ls) npow2 <<= 1;
    const int ngrp = 32 * NHEADS * NHASH;

    dim3 gblk(DMODEL / TN, M / TM);
    dim3 gff (2048 / TN,  M / TM);

    for (int l = 0; l < 2; ++l) {
      launch_gemm(gblk, stream, hbuf, Lb[l].wqk, qkb, M, DMODEL, DMODEL, nullptr, nullptr, 0);
      launch_gemm(gblk, stream, hbuf, Lb[l].wv,  vb,  M, DMODEL, DMODEL, nullptr, nullptr, 0);

      unsigned seed = 0x1234567u + (unsigned)(si * 2 + l) * 2654435761u;
      rot_init_kernel<<<CDIV(NHASH * DH * nb2, 128), 128, 0, stream>>>(rot, NHASH * DH * nb2, seed);
      lsh_bucket_kernel<<<CDIV(ngrp * Ls, 128), 128, 0, stream>>>(qkb, rot, buckets, skey,
                                                                  32, NHEADS, NHASH, Ls, nb2);
      bitonic_sort_kernel<<<ngrp, 256, 0, stream>>>(skey, ticker, undo, Ls, npow2);
      lsh_logits_kernel<<<ngrp * nchunks, 64, 0, stream>>>(qkb, ticker, buckets, logits,
                                                           32, NHEADS, NHASH, Ls, chunk, nchunks);
      merge_weights_kernel<<<CDIV(32 * NHEADS * Ls, 256), 256, 0, stream>>>(logits, undo, wsm,
                                                                            32 * NHEADS * Ls, NHASH, Ls);
      zero_kernel<<<CDIV((long)M * DMODEL, 256), 256, 0, stream>>>(attnb, (long)M * DMODEL);
      lsh_output_kernel<<<ngrp * nchunks, 64, 0, stream>>>(qkb, vb, ticker, buckets, logits, wsm,
                                                           attnb, 32, NHEADS, NHASH, Ls, chunk, nchunks);

      launch_gemm(gblk, stream, attnb, Lb[l].wo, tmp, M, DMODEL, DMODEL, nullptr, hbuf, 0);
      layernorm_kernel<<<M, 256, 0, stream>>>(tmp, Lp[l].ln1g, Lp[l].ln1b, hbuf, DMODEL);

      launch_gemm(gff,  stream, hbuf, Lb[l].ff1, ffmid, M, 2048, DMODEL, Lp[l].ff1b, nullptr, 1);
      launch_gemm(gblk, stream, ffmid, Lb[l].ff2, tmp,  M, DMODEL, 2048, Lp[l].ff2b, hbuf, 0);
      layernorm_kernel<<<M, 256, 0, stream>>>(tmp, Lp[l].ln2g, Lp[l].ln2b, hbuf, DMODEL);
    }

    layernorm_kernel<<<M, 256, 0, stream>>>(hbuf, final_g, final_b, tmp, DMODEL);
    proj_kernel<<<CDIV(M * 7, 256), 256, 0, stream>>>(tmp, proj_w, proj_b, meanb, hproj, 32, Ls);

    copy_rows_kernel<<<CDIV(32 * Ld * 7, 256), 256, 0, stream>>>(hproj, coarse, 32, Ls, Ld, Ls - Ld);
    copy_rows_kernel<<<CDIV(32 * tail * 7, 256), 256, 0, stream>>>(hproj, out + outoff[si], 32, Ls, tail, Ls - tail);
  }
}
